// WordSequence_55722905699235
// MI455X (gfx1250) — matrix operand order repaired and checked by reading, not compiled, not run on hardware
//
#include <hip/hip_runtime.h>
#include <hip/hip_bf16.h>
#include <math.h>

// ---------------------------------------------------------------------------
// Problem constants: VOCAB=30522, D=768, H=128, B=32, S=1024, N_SENT=16,
// SENT_LEN=64. Tokens per stream = 32*1024 = 512*64 = 32768.
// ---------------------------------------------------------------------------
#define DVEC   768
#define HDIM   128
#define G4H    512          // 4*H
#define G2H    256          // 2*H
#define NTOK   32768        // tokens per stream

typedef __attribute__((ext_vector_type(16))) __bf16 v16bf;
typedef __attribute__((ext_vector_type(8)))  __bf16 v8bf;
typedef __attribute__((ext_vector_type(8)))  float  v8f;

__device__ __forceinline__ float sigmf(float x) {
    return 1.0f / (1.0f + __expf(-x));
}

// pack two float4 -> 8 bf16 (lowers to v_cvt_pk_bf16_f32 pairs)
__device__ __forceinline__ v8bf pack8(const float4 a, const float4 b) {
    v8bf p;
    p[0] = (__bf16)a.x; p[1] = (__bf16)a.y; p[2] = (__bf16)a.z; p[3] = (__bf16)a.w;
    p[4] = (__bf16)b.x; p[5] = (__bf16)b.y; p[6] = (__bf16)b.z; p[7] = (__bf16)b.w;
    return p;
}
__device__ __forceinline__ v16bf cat16(v8bf lo, v8bf hi) {
    return __builtin_shufflevector(lo, hi, 0,1,2,3,4,5,6,7,8,9,10,11,12,13,14,15);
}

// A-matrix fragment: lane (half,row) holds k in [16h,16h+16)
// -> one contiguous 32B run per row (matches b_frag k order).
__device__ __forceinline__ v16bf a_frag(const __bf16* rowptr, int half) {
    v8bf lo = *(const v8bf*)(rowptr + 16 * half);
    v8bf hi = *(const v8bf*)(rowptr + 16 * half + 8);
    return cat16(lo, hi);
}
// B-matrix fragment (this code's k order): lane (half,col) holds k in [16h,16h+16)
// -> one contiguous 32B run per column (tile stored column-major in LDS).
__device__ __forceinline__ v16bf b_frag(const __bf16* colptr, int half) {
    v8bf lo = *(const v8bf*)(colptr + 16 * half);
    v8bf hi = *(const v8bf*)(colptr + 16 * half + 8);
    return cat16(lo, hi);
}

// ---------------------------------------------------------------------------
// Kernel 1: fused embedding gather + input projection (pre-activations)
//   pre[tok][g] = sum_d bf16(E[tok_id][d]) * bf16(Wih[g][d]) + b[g]
// grid = (256 M-tiles, 8 N-tiles of 64, 4 projections), block = 256 (8 waves)
// ---------------------------------------------------------------------------
__global__ __launch_bounds__(256) void embed_proj_kernel(
    const int* __restrict__ wtok, const int* __restrict__ stok,
    const float* __restrict__ E,
    const float* __restrict__ W0, const float* __restrict__ b0, float* __restrict__ p0,
    const float* __restrict__ W1, const float* __restrict__ b1, float* __restrict__ p1,
    const float* __restrict__ W2, const float* __restrict__ b2, float* __restrict__ p2,
    const float* __restrict__ W3, const float* __restrict__ b3, float* __restrict__ p3)
{
    const int z = blockIdx.z;
    const float* W    = (z == 0) ? W0 : (z == 1) ? W1 : (z == 2) ? W2 : W3;
    const float* bias = (z == 0) ? b0 : (z == 1) ? b1 : (z == 2) ? b2 : b3;
    float*       pre  = (z == 0) ? p0 : (z == 1) ? p1 : (z == 2) ? p2 : p3;
    const int*   toks = (z < 2) ? wtok : stok;

    const int mbase = blockIdx.x * 128;
    const int nbase = blockIdx.y * 64;
    const int tid   = threadIdx.x;
    const int w     = tid >> 5;
    const int lane  = tid & 31;
    const int half  = lane >> 4;
    const int rl    = lane & 15;

    __shared__ __bf16 aS[128][40];   // 128 x 32 A tile, stride 40 (80B, 16B-aligned rows)
    __shared__ __bf16 bS[64][40];    // B tile COLUMN-major: bS[n][k]
    __shared__ int    tokS[128];

    if (tid < 128) tokS[tid] = toks[mbase + tid];

    v8f acc[4] = {};

    for (int kb = 0; kb < DVEC; kb += 32) {
        __syncthreads();
        // ---- stage A tile: gathered embedding rows, fp32 -> bf16, vector I/O ----
        {
            const int row = tid >> 1;
            const int seg = (tid & 1) * 16;
            const float* src = E + (size_t)tokS[row] * DVEC + kb + seg;
            float4 v0 = *(const float4*)(src + 0);
            float4 v1 = *(const float4*)(src + 4);
            float4 v2 = *(const float4*)(src + 8);
            float4 v3 = *(const float4*)(src + 12);
            *(v8bf*)(&aS[row][seg + 0]) = pack8(v0, v1);
            *(v8bf*)(&aS[row][seg + 8]) = pack8(v2, v3);
            if (kb + 32 < DVEC) __builtin_prefetch(src + 32, 0, 0);  // global_prefetch_b8
        }
        // ---- stage B tile column-major: bS[n][k] = Wih[nbase+n][kb+k] ----
        {
            const int n  = tid & 63;
            const int k0 = (tid >> 6) * 8;
            const float* src = W + (size_t)(nbase + n) * DVEC + kb + k0;
            float4 v0 = *(const float4*)(src + 0);
            float4 v1 = *(const float4*)(src + 4);
            *(v8bf*)(&bS[n][k0]) = pack8(v0, v1);
        }
        __syncthreads();

        const v16bf a = a_frag(&aS[w * 16 + rl][0], half);
        #pragma unroll
        for (int nt = 0; nt < 4; ++nt) {
            const v16bf b = b_frag(&bS[nt * 16 + rl][0], half);
            acc[nt] = __builtin_amdgcn_wmma_f32_16x16x32_bf16(
                false, a, false, b, (short)0, acc[nt], false, false);
        }
    }

    // ---- epilogue: +bias, store fp32 pre-activations (lanes are n-contiguous) ----
    #pragma unroll
    for (int nt = 0; nt < 4; ++nt) {
        const int col = nbase + nt * 16 + rl;
        const float bv = bias[col];
        #pragma unroll
        for (int r = 0; r < 8; ++r) {
            const int m = mbase + w * 16 + (half ? r + 8 : r);
            pre[(size_t)m * G4H + col] = acc[nt][r] + bv;
        }
    }
}

// ---------------------------------------------------------------------------
// Kernel 2: bidirectional LSTM recurrence, WMMA on h @ Whh^T.
// grid = (nb/16 batch tiles, 2 dirs), block = 256 (8 waves).
// Whh B-fragments preloaded into 128 VGPRs (time-invariant). Wave w owns
// hidden units [16w,16w+16): its 4 accumulators are the i/f/g/o gate tiles
// for those units -> lane-local cell update; c lives in VGPRs all scan.
// h ping-pongs through LDS (bf16) with one barrier per step.
// ---------------------------------------------------------------------------
__global__ __launch_bounds__(256) void lstm_rec_kernel(
    const float* __restrict__ preF, const float* __restrict__ preB,
    const float* __restrict__ WhhF, const float* __restrict__ WhhB,
    float* __restrict__ outBuf, int T)
{
    const int dir = blockIdx.y;
    const float* pre = dir ? preB : preF;
    const float* Whh = dir ? WhhB : WhhF;
    const int mbase  = blockIdx.x * 16;

    const int tid  = threadIdx.x;
    const int w    = tid >> 5;
    const int lane = tid & 31;
    const int half = lane >> 4;
    const int rl   = lane & 15;

    __shared__ __bf16 hS[2][16][HDIM + 8];   // stride 136 (272B, 16B-aligned rows)

    // ---- preload Whh^T fragments (contiguous 16-float runs per fragment) ----
    v16bf bf[16];
    #pragma unroll
    for (int kt = 0; kt < 4; ++kt) {
        #pragma unroll
        for (int j = 0; j < 4; ++j) {
            const int n = 128 * j + 16 * w + rl;           // gate output 0..511
            const float* src = Whh + (size_t)n * HDIM + kt * 32 + 16 * half;
            float4 v0 = *(const float4*)(src + 0);
            float4 v1 = *(const float4*)(src + 4);
            float4 v2 = *(const float4*)(src + 8);
            float4 v3 = *(const float4*)(src + 12);
            bf[kt * 4 + j] = cat16(pack8(v0, v1), pack8(v2, v3));
        }
    }

    // ---- zero h ping-pong buffers + cell state ----
    for (int i = tid; i < 2 * 16 * (HDIM + 8); i += 256)
        ((__bf16*)hS)[i] = (__bf16)0.0f;
    float creg[8];
    #pragma unroll
    for (int r = 0; r < 8; ++r) creg[r] = 0.0f;
    __syncthreads();

    const int u = 16 * w + rl;   // hidden unit owned by this lane

    for (int s = 0; s < T; ++s) {
        const int t   = dir ? (T - 1 - s) : s;
        const int cur = s & 1;
        const int nxt = cur ^ 1;

        // ---- init accumulators from precomputed input projection ----
        v8f acc[4];
        #pragma unroll
        for (int j = 0; j < 4; ++j) {
            const int n = 128 * j + 16 * w + rl;
            #pragma unroll
            for (int r = 0; r < 8; ++r) {
                const int rowl = half ? r + 8 : r;
                acc[j][r] = pre[((size_t)(mbase + rowl) * T + t) * G4H + n];
            }
        }

        // ---- g += h_{t-1} @ Whh^T : 16 WMMAs, A frag = 2x ds_load_b128 ----
        #pragma unroll
        for (int kt = 0; kt < 4; ++kt) {
            const v16bf a = a_frag(&hS[cur][rl][kt * 32], half);
            #pragma unroll
            for (int j = 0; j < 4; ++j)
                acc[j] = __builtin_amdgcn_wmma_f32_16x16x32_bf16(
                    false, a, false, bf[kt * 4 + j], (short)0, acc[j], false, false);
        }

        // ---- lane-local LSTM cell update (i,f,g,o gate order) ----
        #pragma unroll
        for (int r = 0; r < 8; ++r) {
            const int rowl = half ? r + 8 : r;
            const float iv = acc[0][r];
            const float fv = acc[1][r];
            const float gv = acc[2][r];
            const float ov = acc[3][r];
            const float cc = sigmf(fv) * creg[r] + sigmf(iv) * tanhf(gv);
            const float hh = sigmf(ov) * tanhf(cc);
            creg[r] = cc;
            hS[nxt][rowl][u] = (__bf16)hh;
            outBuf[((size_t)(mbase + rowl) * T + t) * G2H + dir * HDIM + u] = hh;
        }
        __syncthreads();
    }
}

// ---------------------------------------------------------------------------
// Kernel 3: gamma = sigmoid([fw|fs] @ gate_W^T + gate_b); out = mix(fs,fw,gamma)
// M=32768, N=256, K=512. grid = (256, 4), block = 256.
// ---------------------------------------------------------------------------
__global__ __launch_bounds__(256) void gate_blend_kernel(
    const float* __restrict__ fw, const float* __restrict__ fs,
    const float* __restrict__ gW, const float* __restrict__ gb,
    float* __restrict__ out)
{
    const int mbase = blockIdx.x * 128;
    const int nbase = blockIdx.y * 64;
    const int tid   = threadIdx.x;
    const int w     = tid >> 5;
    const int lane  = tid & 31;
    const int half  = lane >> 4;
    const int rl    = lane & 15;

    __shared__ __bf16 aS[128][40];
    __shared__ __bf16 bS[64][40];    // column-major: bS[n][k]

    v8f acc[4] = {};

    for (int kb = 0; kb < G4H; kb += 32) {
        __syncthreads();
        // ---- stage A: concat[fw | fs] rows -> bf16 ----
        {
            const int row = tid >> 1;
            const int seg = (tid & 1) * 16;
            const int m   = mbase + row;
            const int kg  = kb + seg;          // 32-aligned chunk never straddles 256
            const float* src = (kg < G2H) ? (fw + (size_t)m * G2H + kg)
                                          : (fs + (size_t)m * G2H + (kg - G2H));
            float4 v0 = *(const float4*)(src + 0);
            float4 v1 = *(const float4*)(src + 4);
            float4 v2 = *(const float4*)(src + 8);
            float4 v3 = *(const float4*)(src + 12);
            *(v8bf*)(&aS[row][seg + 0]) = pack8(v0, v1);
            *(v8bf*)(&aS[row][seg + 8]) = pack8(v2, v3);
        }
        // ---- stage B column-major: bS[n][k] = gate_W[nbase+n][kb+k] ----
        {
            const int n  = tid & 63;
            const int k0 = (tid >> 6) * 8;
            const float* src = gW + (size_t)(nbase + n) * G4H + kb + k0;
            float4 v0 = *(const float4*)(src + 0);
            float4 v1 = *(const float4*)(src + 4);
            *(v8bf*)(&bS[n][k0]) = pack8(v0, v1);
        }
        __syncthreads();

        const v16bf a = a_frag(&aS[w * 16 + rl][0], half);
        #pragma unroll
        for (int nt = 0; nt < 4; ++nt) {
            const v16bf b = b_frag(&bS[nt * 16 + rl][0], half);
            acc[nt] = __builtin_amdgcn_wmma_f32_16x16x32_bf16(
                false, a, false, b, (short)0, acc[nt], false, false);
        }
    }

    // ---- epilogue: sigmoid gate + convex blend ----
    #pragma unroll
    for (int nt = 0; nt < 4; ++nt) {
        const int col = nbase + nt * 16 + rl;
        const float bv = gb[col];
        #pragma unroll
        for (int r = 0; r < 8; ++r) {
            const int m = mbase + w * 16 + (half ? r + 8 : r);
            const float gamma = sigmf(acc[nt][r] + bv);
            const float fwv = fw[(size_t)m * G2H + col];
            const float fsv = fs[(size_t)m * G2H + col];
            out[(size_t)m * G2H + col] = gamma * fwv + (1.0f - gamma) * fsv;
        }
    }
}

// ---------------------------------------------------------------------------
// Launch
// ---------------------------------------------------------------------------
extern "C" void kernel_launch(void* const* d_in, const int* in_sizes, int n_in,
                              void* d_out, int out_size, void* d_ws, size_t ws_size,
                              hipStream_t stream) {
    const int*   wtok   = (const int*)d_in[0];     // [32,1024]
    const int*   stok   = (const int*)d_in[1];     // [32,16,64] (flat 32768)
    const float* E      = (const float*)d_in[3];   // [30522,768]
    const float* lWih_f = (const float*)d_in[4];   // [512,768]
    const float* lWhh_f = (const float*)d_in[5];   // [512,128]
    const float* lb_f   = (const float*)d_in[6];   // [512]
    const float* lWih_b = (const float*)d_in[7];
    const float* lWhh_b = (const float*)d_in[8];
    const float* lb_b   = (const float*)d_in[9];
    const float* sWih_f = (const float*)d_in[10];
    const float* sWhh_f = (const float*)d_in[11];
    const float* sb_f   = (const float*)d_in[12];
    const float* sWih_b = (const float*)d_in[13];
    const float* sWhh_b = (const float*)d_in[14];
    const float* sb_b   = (const float*)d_in[15];
    const float* gW     = (const float*)d_in[16];  // [256,512]
    const float* gb     = (const float*)d_in[17];  // [256]
    float* out = (float*)d_out;                    // [32,1024,256]

    const size_t PRE = (size_t)NTOK * G4H;
    const size_t FO  = (size_t)NTOK * G2H;
    float* ws    = (float*)d_ws;
    float* preWF = ws;
    float* preWB = ws + 1 * PRE;
    float* preSF = ws + 2 * PRE;
    float* preSB = ws + 3 * PRE;
    float* fw    = ws + 4 * PRE;
    float* fs    = fw + FO;

    embed_proj_kernel<<<dim3(NTOK / 128, G4H / 64, 4), 256, 0, stream>>>(
        wtok, stok, E,
        lWih_f, lb_f, preWF,
        lWih_b, lb_b, preWB,
        sWih_f, sb_f, preSF,
        sWih_b, sb_b, preSB);

    lstm_rec_kernel<<<dim3(2, 2), 256, 0, stream>>>(
        preWF, preWB, lWhh_f, lWhh_b, fw, 1024);

    lstm_rec_kernel<<<dim3(32, 2), 256, 0, stream>>>(
        preSF, preSB, sWhh_f, sWhh_b, fs, 64);

    gate_blend_kernel<<<dim3(NTOK / 128, G2H / 64), 256, 0, stream>>>(
        fw, fs, gW, gb, out);
}
